// EquivariantAtomEncoder_34797825032823
// MI455X (gfx1250) — matrix operand order repaired and checked by reading, not compiled, not run on hardware
//
#include <hip/hip_runtime.h>
#include <hip/hip_bf16.h>

// ---------------------------------------------------------------------------
// EquivariantAtomEncoder for MI455X (gfx1250, wave32, WMMA)
//
// B=32 N=64 -> BN=2048 nodes, E=131072 edges
// CS=64 CV=32 RBF=32 HID=128 NW=224 NBLK=3
//
// Heavy compute = per-edge radial MLP: (Ex32)@(32x128) then (Ex128)@(128x224),
// mapped to v_wmma_f32_16x16x32_f16 with 16-edge tiles per wave.
// GEMM2 output is consumed in-register -> message TP -> float atomics into
// per-node aggregation buffers (fit in L2), never written to HBM.
// All divisions use v_rcp_f32 (activation funcs tolerate 1-ulp rcp).
// ---------------------------------------------------------------------------

typedef __attribute__((ext_vector_type(16))) _Float16 v16h;
typedef __attribute__((ext_vector_type(8)))  _Float16 v8h;
typedef __attribute__((ext_vector_type(8)))  float    v8f;

#define EDGES   131072
#define NODES   2048
#define CUTOFF_F 5.0f
#define PI_F     3.14159265358979f

__device__ __forceinline__ float fast_rcp(float x)     { return __builtin_amdgcn_rcpf(x); }
__device__ __forceinline__ float fast_sigmoid(float x) { return fast_rcp(1.0f + __expf(-x)); }
__device__ __forceinline__ float fast_silu(float x)    { return x * fast_sigmoid(x); }

// ---------------------------------------------------------------------------
// Kernel 0: convert W1 (3,32,128) and W2 (3,128,224) to f16 in WMMA B-layout.
// B-layout (16-bit, 32x16 KxN): lane l -> N = l&15, halves h<8 -> K = (l<16?0:8)+h;
// h>=8 -> K = (l<16?16:24)+h-8.   Tile storage: [...][lane][h] contiguous.
// ---------------------------------------------------------------------------
__global__ __launch_bounds__(256)
void prep_weights(const float* __restrict__ W1, const float* __restrict__ W2,
                  _Float16* __restrict__ B1w, _Float16* __restrict__ B2w) {
    int t = blockIdx.x * blockDim.x + threadIdx.x;
    if (t < 3 * 8 * 512) {              // W1: 3 blk x 8 n-tiles x (32 lanes x 16 halves)
        int blk = t / 4096;
        int r   = t % 4096;
        int nt  = r >> 9;
        int li  = (r & 511) >> 4;
        int h   = r & 15;
        int K   = (h < 8) ? (((li < 16) ? 0 : 8) + h) : (((li < 16) ? 16 : 24) + h - 8);
        int n   = nt * 16 + (li & 15);
        B1w[t] = (_Float16)W1[(blk * 32 + K) * 128 + n];
    }
    int u = t - 12288;
    if (u >= 0 && u < 3 * 4 * 14 * 512) {   // W2: 3 blk x 4 k-tiles x 14 n-tiles
        int blk = u / 28672;
        int r   = u % 28672;
        int kt  = r / 7168;
        int r2  = r % 7168;
        int nt  = r2 >> 9;
        int li  = (r2 & 511) >> 4;
        int h   = r2 & 15;
        int K   = kt * 32 + ((h < 8) ? (((li < 16) ? 0 : 8) + h) : (((li < 16) ? 16 : 24) + h - 8));
        int n   = nt * 16 + (li & 15);
        B2w[u] = (_Float16)W2[(blk * 128 + K) * 224 + n];
    }
}

// ---------------------------------------------------------------------------
// Kernel 1: per-edge geometry features: Y1 = sqrt(3)*unit(edge_vec), cosine env
// ---------------------------------------------------------------------------
__global__ __launch_bounds__(256)
void edge_feat(const float* __restrict__ ew, const float* __restrict__ ev,
               float* __restrict__ Y1, float* __restrict__ env) {
    int e = blockIdx.x * blockDim.x + threadIdx.x;
    if (e >= EDGES) return;
    float el  = ew[e];
    float inv = fast_rcp(fmaxf(el, 1e-8f));
    const float s3 = 1.7320508075688772f;
    Y1[e * 3 + 0] = s3 * ev[e * 3 + 0] * inv;
    Y1[e * 3 + 1] = s3 * ev[e * 3 + 1] * inv;
    Y1[e * 3 + 2] = s3 * ev[e * 3 + 2] * inv;
    float en = 0.5f * (__cosf(PI_F * el * (1.0f / CUTOFF_F)) + 1.0f);
    env[e] = (el < CUTOFF_F) ? en : 0.0f;
}

// ---------------------------------------------------------------------------
// Kernel 2: Gaussian RBF, written directly in WMMA A-layout (f16, 16x32 MxK):
// lane l -> M(edge) = l&15; halves h<8 -> K = (l<16?0:8)+h ; h>=8 -> K = (l<16?16:24)+h-8
// Tile storage: rbfA[tile][lane][h].
// ---------------------------------------------------------------------------
__global__ __launch_bounds__(256)
void rbf_layout(const float* __restrict__ ew, _Float16* __restrict__ rbfA) {
    int t = blockIdx.x * blockDim.x + threadIdx.x;
    const int tiles = EDGES / 16;
    if (t >= tiles * 32) return;
    int tile = t >> 5;
    int l    = t & 31;
    int e    = tile * 16 + (l & 15);
    float d  = fminf(ew[e], CUTOFF_F);
    const float delta     = CUTOFF_F / 31.0f;
    const float inv_delta = 31.0f / CUTOFF_F;
    v16h out;
#pragma unroll
    for (int h = 0; h < 16; ++h) {
        int K = (h < 8) ? (((l < 16) ? 0 : 8) + h) : (((l < 16) ? 16 : 24) + h - 8);
        float x = (d - (float)K * delta) * inv_delta;
        out[h] = (_Float16)__expf(-0.5f * x * x);
    }
    *(v16h*)(rbfA + (size_t)tile * 512 + l * 16) = out;
}

// ---------------------------------------------------------------------------
// Kernel 3: node init: s = [z_emb[z], abs_flag] @ W_in (65x64), masked; v = 0
// ---------------------------------------------------------------------------
__global__ __launch_bounds__(64)
void init_nodes(const int* __restrict__ z, const float* __restrict__ mask,
                const int* __restrict__ absi, const float* __restrict__ z_emb,
                const float* __restrict__ W_in,
                float* __restrict__ sbuf, float* __restrict__ vbuf) {
    int n = blockIdx.x, c = threadIdx.x;
    int zi = z[n];
    float fm = mask[n];
    int b = n >> 6, i = n & 63;
    float flag = (absi[b] == i) ? 1.0f : 0.0f;
    float acc = flag * W_in[64 * 64 + c];          // row 64 = abs_flag row
    for (int a = 0; a < 64; ++a)
        acc += z_emb[zi * 64 + a] * W_in[a * 64 + c];
    sbuf[n * 64 + c] = acc * fm;
    vbuf[n * 96 + c] = 0.0f;
    if (c < 32) vbuf[n * 96 + 64 + c] = 0.0f;
}

// ---------------------------------------------------------------------------
// Kernel 4 (main): fused edge MLP (two WMMA GEMMs) + TP messages + scatter.
// 4 waves / workgroup, 16 edges / wave, grid = E/64.
// ---------------------------------------------------------------------------
__global__ __launch_bounds__(128)
void edge_gemm_scatter(const _Float16* __restrict__ rbfA,
                       const _Float16* __restrict__ B1w,
                       const _Float16* __restrict__ B2w,
                       const float* __restrict__ b1g, const float* __restrict__ b2g,
                       const float* __restrict__ envg, const float* __restrict__ Y1g,
                       const int* __restrict__ esrc, const int* __restrict__ edst,
                       const float* __restrict__ sbuf, const float* __restrict__ vbuf,
                       float* __restrict__ agg_s, float* __restrict__ agg_v, int blk) {
    __shared__ __align__(32) _Float16 hbuf[4][16 * 128];  // h tiles, one per wave
    __shared__ int   s_src[4][16], s_dst[4][16];
    __shared__ float s_env[4][16], s_y[4][16][3];

    const int lane = threadIdx.x & 31;
    const int wv   = threadIdx.x >> 5;
    const int tile = blockIdx.x * 4 + wv;
    const int base = tile * 16;
    const int col16 = lane & 15;
    const int hi    = lane >> 4;

    // stage per-edge metadata for this wave's 16 edges
    if (lane < 16) {
        s_src[wv][lane] = esrc[base + lane];
        s_dst[wv][lane] = edst[base + lane];
        s_env[wv][lane] = envg[base + lane];
    } else {
        int e = lane - 16;
        s_y[wv][e][0] = Y1g[(base + e) * 3 + 0];
        s_y[wv][e][1] = Y1g[(base + e) * 3 + 1];
        s_y[wv][e][2] = Y1g[(base + e) * 3 + 2];
    }

    // A fragment for GEMM1: pre-laid-out RBF tile, one 32B load per lane
    v16h a1 = *(const v16h*)(rbfA + (size_t)tile * 512 + lane * 16);

    __syncthreads();

    // ---- GEMM1: h = silu(rbf @ W1 + b1), N=128 in 8 tiles of 16 -----------
#pragma unroll
    for (int nt = 0; nt < 8; ++nt) {
        float bias = b1g[blk * 128 + nt * 16 + col16];
        v8f c;
#pragma unroll
        for (int i = 0; i < 8; ++i) c[i] = bias;
        v16h bfrag = *(const v16h*)(B1w + (size_t)(blk * 8 + nt) * 512 + lane * 16);
        c = __builtin_amdgcn_wmma_f32_16x16x32_f16(false, a1, false, bfrag,
                                                   (short)0, c, false, false);
#pragma unroll
        for (int r = 0; r < 8; ++r) {                 // C-layout: M = r + 8*hi
            float hs = fast_silu(c[r]);
            int edge = r + 8 * hi;
            hbuf[wv][edge * 128 + nt * 16 + col16] = (_Float16)hs;
        }
    }

    __syncthreads();

    // ---- GEMM2 + env scale + TP messages + scatter-add --------------------
    const float inv_s3 = 0.5773502691896258f;   // 1/sqrt(3)
    const float inv_s2 = 0.7071067811865476f;   // 1/sqrt(2)
    for (int nt = 0; nt < 14; ++nt) {
        float bias = b2g[blk * 224 + nt * 16 + col16];
        v8f acc;
#pragma unroll
        for (int i = 0; i < 8; ++i) acc[i] = bias;
#pragma unroll
        for (int kt = 0; kt < 4; ++kt) {
            // rebuild A fragment of h from LDS (A-layout addressing)
            int kb1 = kt * 32 + (hi ? 8 : 0);
            int kb2 = kt * 32 + (hi ? 24 : 16);
            v8h lo = *(const v8h*)&hbuf[wv][col16 * 128 + kb1];
            v8h hh = *(const v8h*)&hbuf[wv][col16 * 128 + kb2];
            v16h af;
#pragma unroll
            for (int i = 0; i < 8; ++i) { af[i] = lo[i]; af[i + 8] = hh[i]; }
            v16h bfrag = *(const v16h*)(B2w + (size_t)((blk * 4 + kt) * 14 + nt) * 512 + lane * 16);
            acc = __builtin_amdgcn_wmma_f32_16x16x32_f16(false, af, false, bfrag,
                                                         (short)0, acc, false, false);
        }
        // channel column; boundaries (64/96/160/192) are multiples of 16 so the
        // branch below is uniform across the wave for each nt
        int col = nt * 16 + col16;
#pragma unroll
        for (int r = 0; r < 8; ++r) {
            int e = r + 8 * hi;
            float w = acc[r] * s_env[wv][e];
            int src = s_src[wv][e], dst = s_dst[wv][e];
            float y0 = s_y[wv][e][0], y1 = s_y[wv][e][1], y2 = s_y[wv][e][2];
            if (col < 64) {                               // w_ss: 0e x 0e -> 0e
                float ss = sbuf[src * 64 + col];
                atomicAdd(&agg_s[dst * 96 + col], w * ss);
            } else if (col < 96) {                        // w_vv: 1o x 1o -> 0e
                int cv = col - 64;
                const float* vp = vbuf + (size_t)(src * 32 + cv) * 3;
                float dv = vp[0] * y0 + vp[1] * y1 + vp[2] * y2;
                atomicAdd(&agg_s[dst * 96 + col], w * dv * inv_s3);
            } else if (col < 160) {                       // w_sv: 0e x 1o -> 1o
                int ch = col - 96;
                float ss = sbuf[src * 64 + ch];
                float* ap = agg_v + (size_t)(dst * 128 + ch) * 3;
                atomicAdd(ap + 0, w * ss * y0);
                atomicAdd(ap + 1, w * ss * y1);
                atomicAdd(ap + 2, w * ss * y2);
            } else if (col < 192) {                       // w_vs: 1o x 0e -> 1o
                int cv = col - 160;
                const float* vp = vbuf + (size_t)(src * 32 + cv) * 3;
                float* ap = agg_v + (size_t)(dst * 128 + 64 + cv) * 3;
                atomicAdd(ap + 0, w * vp[0]);
                atomicAdd(ap + 1, w * vp[1]);
                atomicAdd(ap + 2, w * vp[2]);
            } else {                                      // w_vx: cross -> 1o
                int cv = col - 192;
                const float* vp = vbuf + (size_t)(src * 32 + cv) * 3;
                float cx = vp[1] * y2 - vp[2] * y1;
                float cy = vp[2] * y0 - vp[0] * y2;
                float cz = vp[0] * y1 - vp[1] * y0;
                float* ap = agg_v + (size_t)(dst * 128 + 96 + cv) * 3;
                atomicAdd(ap + 0, w * cx * inv_s2);
                atomicAdd(ap + 1, w * cy * inv_s2);
                atomicAdd(ap + 2, w * cz * inv_s2);
            }
        }
    }
}

// ---------------------------------------------------------------------------
// Kernel 5: node update: us = silu(agg_s @ Ws); uv = agg_v @ Wv;
// g = sigmoid(us @ Wg); s += a*us; v += a*g*uv. One block per node.
// ---------------------------------------------------------------------------
__global__ __launch_bounds__(128)
void node_update(const float* __restrict__ agg_s, const float* __restrict__ agg_v,
                 const float* __restrict__ Ws, const float* __restrict__ Wv,
                 const float* __restrict__ Wg, const float* __restrict__ alpha,
                 float* __restrict__ sbuf, float* __restrict__ vbuf, int blk) {
    __shared__ float us_sh[64];
    int n = blockIdx.x, t = threadIdx.x;
    float al = alpha[blk];
    if (t < 64) {
        float acc = 0.0f;
        for (int j = 0; j < 96; ++j)
            acc += agg_s[n * 96 + j] * Ws[(blk * 96 + j) * 64 + t];
        float us = fast_silu(acc);
        us_sh[t] = us;
        sbuf[n * 64 + t] += al * us;
    }
    __syncthreads();
    if (t < 96) {
        int cv = t / 3, d = t - cv * 3;
        float uv = 0.0f;
        for (int j = 0; j < 128; ++j)
            uv += agg_v[(size_t)(n * 128 + j) * 3 + d] * Wv[(blk * 128 + j) * 32 + cv];
        float ga = 0.0f;
        for (int c = 0; c < 64; ++c)
            ga += us_sh[c] * Wg[(blk * 64 + c) * 32 + cv];
        float g = fast_sigmoid(ga);
        vbuf[(size_t)(n * 32 + cv) * 3 + d] += al * g * uv;
    }
}

// ---------------------------------------------------------------------------
// Kernel 6: IrrepNorm + mask + pack output (BN x 160)
// ---------------------------------------------------------------------------
__global__ __launch_bounds__(128)
void finalize(const float* __restrict__ sbuf, const float* __restrict__ vbuf,
              const float* __restrict__ mask, float* __restrict__ out) {
    __shared__ float rs[128], rv[128];
    int n = blockIdx.x, t = threadIdx.x;
    float sq = 0.0f, vq = 0.0f;
    if (t < 64) { float x = sbuf[n * 64 + t]; sq = x * x; }
    if (t < 32) {
        const float* vp = vbuf + (size_t)(n * 32 + t) * 3;
        vq = vp[0] * vp[0] + vp[1] * vp[1] + vp[2] * vp[2];
    }
    rs[t] = sq; rv[t] = vq;
    for (int off = 64; off >= 1; off >>= 1) {
        __syncthreads();
        if (t < off) { rs[t] += rs[t + off]; rv[t] += rv[t + off]; }
    }
    __syncthreads();
    float fm = mask[n];
    float sinv = fm * rsqrtf(rs[0] * (1.0f / 64.0f) + 1e-6f);
    float vinv = fm * rsqrtf(rv[0] * (1.0f / 32.0f) + 1e-6f);
    if (t < 64) out[n * 160 + t]      = sbuf[n * 64 + t] * sinv;
    if (t < 96) out[n * 160 + 64 + t] = vbuf[n * 96 + t] * vinv;
}

// ---------------------------------------------------------------------------
extern "C" void kernel_launch(void* const* d_in, const int* in_sizes, int n_in,
                              void* d_out, int out_size, void* d_ws, size_t ws_size,
                              hipStream_t stream) {
    (void)in_sizes; (void)n_in; (void)out_size; (void)ws_size;
    const int*   z     = (const int*)d_in[0];
    const float* mask  = (const float*)d_in[1];
    const int*   absi  = (const int*)d_in[2];
    const int*   esrc  = (const int*)d_in[3];
    const int*   edst  = (const int*)d_in[4];
    const float* ew    = (const float*)d_in[5];
    const float* ev    = (const float*)d_in[6];
    const float* z_emb = (const float*)d_in[7];
    const float* W_in  = (const float*)d_in[8];
    const float* W1    = (const float*)d_in[9];
    const float* b1    = (const float*)d_in[10];
    const float* W2    = (const float*)d_in[11];
    const float* b2    = (const float*)d_in[12];
    const float* Ws    = (const float*)d_in[13];
    const float* Wv    = (const float*)d_in[14];
    const float* Wg    = (const float*)d_in[15];
    const float* alpha = (const float*)d_in[16];
    float* out = (float*)d_out;

    char* ws = (char*)d_ws;
    size_t off = 0;
    float* sbuf  = (float*)(ws + off); off += (size_t)NODES * 64 * 4;        // 512 KB
    float* vbuf  = (float*)(ws + off); off += (size_t)NODES * 96 * 4;        // 768 KB
    float* aggs  = (float*)(ws + off); off += (size_t)NODES * 96 * 4;        // 768 KB
    float* aggv  = (float*)(ws + off); off += (size_t)NODES * 384 * 4;       // 3 MB
    float* Y1    = (float*)(ws + off); off += (size_t)EDGES * 3 * 4;         // 1.5 MB
    float* env   = (float*)(ws + off); off += (size_t)EDGES * 4;             // 0.5 MB
    _Float16* rbfA = (_Float16*)(ws + off); off += (size_t)(EDGES / 16) * 512 * 2; // 8 MB
    _Float16* B1w  = (_Float16*)(ws + off); off += 12288 * 2;
    _Float16* B2w  = (_Float16*)(ws + off); off += 86016 * 2;

    // --- preprocessing -----------------------------------------------------
    prep_weights<<<(12288 + 86016 + 255) / 256, 256, 0, stream>>>(W1, W2, B1w, B2w);
    edge_feat<<<EDGES / 256, 256, 0, stream>>>(ew, ev, Y1, env);
    rbf_layout<<<(EDGES / 16) * 32 / 256, 256, 0, stream>>>(ew, rbfA);
    init_nodes<<<NODES, 64, 0, stream>>>(z, mask, absi, z_emb, W_in, sbuf, vbuf);

    // --- 3 message-passing blocks -----------------------------------------
    for (int blk = 0; blk < 3; ++blk) {
        hipMemsetAsync(aggs, 0, (size_t)NODES * 96 * 4, stream);
        hipMemsetAsync(aggv, 0, (size_t)NODES * 384 * 4, stream);
        edge_gemm_scatter<<<EDGES / 64, 128, 0, stream>>>(
            rbfA, B1w, B2w, b1, b2, env, Y1, esrc, edst, sbuf, vbuf, aggs, aggv, blk);
        node_update<<<NODES, 128, 0, stream>>>(aggs, aggv, Ws, Wv, Wg, alpha,
                                               sbuf, vbuf, blk);
    }

    // --- norm + output ------------------------------------------------------
    finalize<<<NODES, 128, 0, stream>>>(sbuf, vbuf, mask, out);
}
